// GraphBlock_31344671326364
// MI455X (gfx1250) — matrix operand order repaired and checked by reading, not compiled, not run on hardware
//
#include <hip/hip_runtime.h>
#include <stdint.h>

typedef __attribute__((ext_vector_type(16))) __bf16 v16bf;
typedef __attribute__((ext_vector_type(8)))  __bf16 bf16x8;
typedef __attribute__((ext_vector_type(4)))  __bf16 bf16x4;
typedef __attribute__((ext_vector_type(8)))  float  v8f;
typedef __attribute__((ext_vector_type(4)))  float  f32x4;

#define B_ 8
#define N_ 2048
#define F_ 128

__device__ __forceinline__ unsigned short f2bf(float f) {
  __bf16 h = (__bf16)f;                    // native v_cvt (RNE)
  return __builtin_bit_cast(unsigned short, h);
}

union FragBF { v16bf v; uint4 q[2]; };

__device__ __forceinline__ void cvt8(unsigned short* dst, f32x4 p0, f32x4 p1) {
  bf16x4 a = __builtin_convertvector(p0, bf16x4);   // v_cvt_pk_bf16_f32 x2
  bf16x4 b = __builtin_convertvector(p1, bf16x4);
  bf16x8 o = __builtin_shufflevector(a, b, 0, 1, 2, 3, 4, 5, 6, 7);
  *(bf16x8*)dst = o;                                // single 16B LDS store
}

// ---------------------------------------------------------------------------
// Kernel 1: xw = x @ W (bf16 WMMA, fp32 accum), stored TRANSPOSED as
// xwT[b][f][n] bf16 so GEMM2 B-fragments are contiguous 32B runs.
// ---------------------------------------------------------------------------
__global__ __launch_bounds__(256) void k_xw(const float* __restrict__ x,
                                            const float* __restrict__ W,
                                            unsigned short* __restrict__ xwT) {
  __shared__ unsigned short sWT[F_][136];
  __shared__ unsigned short sX[64][40];
  const int t    = threadIdx.x;
  const int b    = blockIdx.x >> 5;
  const int n0   = (blockIdx.x & 31) << 6;
  const int lane = t & 31, w = t >> 5;
  const int mb   = w & 3,  fh = w >> 2;
  const int lhi  = lane >> 4, lmod = lane & 15;

  for (int i = t; i < F_ * F_; i += 256) {
    int k = i >> 7, f = i & 127;
    sWT[f][k] = f2bf(W[i]);
  }

  v8f acc[4] = {};
  const int arow = t >> 2, aseg = t & 3;

  for (int kt = 0; kt < F_; kt += 32) {
    __syncthreads();
    {
      const f32x4* src =
          (const f32x4*)(x + ((size_t)(b * N_ + n0 + arow)) * F_ + kt + aseg * 8);
      cvt8(&sX[arow][aseg * 8], src[0], src[1]);
    }
    __syncthreads();
    FragBF fa;
    const unsigned short* ap = &sX[mb * 16 + lmod][lhi * 16];
    fa.q[0] = *(const uint4*)ap;
    fa.q[1] = *(const uint4*)(ap + 8);
    FragBF fb[4];
#pragma unroll
    for (int j = 0; j < 4; ++j) {
      const unsigned short* bp = &sWT[fh * 64 + j * 16 + lmod][kt + lhi * 16];
      fb[j].q[0] = *(const uint4*)bp;
      fb[j].q[1] = *(const uint4*)(bp + 8);
    }
#pragma unroll
    for (int j = 0; j < 4; ++j)
      acc[j] = __builtin_amdgcn_wmma_f32_16x16x32_bf16(
          false, fa.v, false, fb[j].v, (short)0, acc[j], false, false);
  }
#pragma unroll
  for (int j = 0; j < 4; ++j) {             // 8 outputs are contiguous in xwT:
    const int f = fh * 64 + j * 16 + lmod;  // one bf16x8 (16B) store each
    size_t base = ((size_t)b * F_ + f) * N_ + n0 + mb * 16 + lhi * 8;
    *(bf16x8*)(xwT + base) = __builtin_convertvector(acc[j], bf16x8);
  }
}

// ---------------------------------------------------------------------------
// Kernel 2: h = a @ xw (8.6 GFLOP). Double-buffered LDS: next tile's global
// loads (a as NT — 134 MB streamed once, keeps L2 for reused xwT/h) issue
// before the barrier and retire into buffer cur^1 while WMMAs consume buffer
// cur. Per-channel Σ/Σx² reduced in the epilogue (ds_add_f32 + one global
// atomic per (b,f) per WG) so h is never re-read for stats.
// ---------------------------------------------------------------------------
__global__ __launch_bounds__(256) void k_gcn(const float* __restrict__ a,
                                             const unsigned short* __restrict__ xwT,
                                             float* __restrict__ h,
                                             float* __restrict__ gsum,
                                             float* __restrict__ gsq) {
  __shared__ unsigned short sA[2][64][40];
  __shared__ unsigned short sB[2][F_][40];
  __shared__ float sStat[2][F_];
  const int t    = threadIdx.x;
  const int b    = blockIdx.x >> 5;
  const int n0   = (blockIdx.x & 31) << 6;
  const int lane = t & 31, w = t >> 5;
  const int mb   = w & 3,  fh = w >> 2;
  const int lhi  = lane >> 4, lmod = lane & 15;
  const int arow = t >> 2, aseg = t & 3;    // 4 threads / a-row, 8 floats each
  const int bfr  = t >> 1, bseg = t & 1;    // 2 threads / xwT-row, 16 bf16 each

  if (t < F_) { sStat[0][t] = 0.f; sStat[1][t] = 0.f; }

  const float* aRow = a + ((size_t)b * N_ + n0 + arow) * N_;
  const f32x4* aSeg = (const f32x4*)(aRow + aseg * 8);
  const uint4* bRow = (const uint4*)(xwT + ((size_t)b * F_ + bfr) * N_);

  v8f acc[4] = {};

  // prologue: stage tile kt=0 into buffer 0
  f32x4 ap0 = __builtin_nontemporal_load(aSeg + 0);
  f32x4 ap1 = __builtin_nontemporal_load(aSeg + 1);
  uint4 bq0 = bRow[bseg * 2];
  uint4 bq1 = bRow[bseg * 2 + 1];
  cvt8(&sA[0][arow][aseg * 8], ap0, ap1);
  { uint4* bd = (uint4*)&sB[0][bfr][bseg * 16]; bd[0] = bq0; bd[1] = bq1; }

  int cur = 0;
  for (int kt = 0; kt < N_; kt += 32) {
    const bool more = (kt + 32) < N_;
    if (more) {   // issue next tile's global loads before the barrier
      const f32x4* s = aSeg + ((kt + 32) >> 2);
      ap0 = __builtin_nontemporal_load(s + 0);
      ap1 = __builtin_nontemporal_load(s + 1);
      const int ko = (kt + 32 + bseg * 16) >> 3;
      bq0 = bRow[ko];
      bq1 = bRow[ko + 1];
    }
    __syncthreads();                         // buffer cur is ready
    FragBF fa;
    const unsigned short* ap = &sA[cur][mb * 16 + lmod][lhi * 16];
    fa.q[0] = *(const uint4*)ap;
    fa.q[1] = *(const uint4*)(ap + 8);
    FragBF fb[4];
#pragma unroll
    for (int j = 0; j < 4; ++j) {
      const unsigned short* bp = &sB[cur][fh * 64 + j * 16 + lmod][lhi * 16];
      fb[j].q[0] = *(const uint4*)bp;
      fb[j].q[1] = *(const uint4*)(bp + 8);
    }
#pragma unroll
    for (int j = 0; j < 4; ++j)
      acc[j] = __builtin_amdgcn_wmma_f32_16x16x32_bf16(
          false, fa.v, false, fb[j].v, (short)0, acc[j], false, false);
    if (more) {   // deposit next tile into the other buffer (no extra barrier)
      cvt8(&sA[cur ^ 1][arow][aseg * 8], ap0, ap1);
      uint4* bd = (uint4*)&sB[cur ^ 1][bfr][bseg * 16];
      bd[0] = bq0; bd[1] = bq1;
    }
    cur ^= 1;
  }

  // epilogue: store h tile + fused per-channel partial stats
#pragma unroll
  for (int j = 0; j < 4; ++j) {
    const int f = fh * 64 + j * 16 + lmod;
    const size_t node = (size_t)n0 + mb * 16 + lhi * 8;
    float s = 0.f, q = 0.f;
#pragma unroll
    for (int r = 0; r < 8; ++r) {
      float v = acc[j][r];
      h[((size_t)b * N_ + node + r) * F_ + f] = v;
      s += v; q += v * v;
    }
    atomicAdd(&sStat[0][f], s);
    atomicAdd(&sStat[1][f], q);
  }
  __syncthreads();
  if (t < F_) {
    atomicAdd(&gsum[b * F_ + t], sStat[0][t]);
    atomicAdd(&gsq [b * F_ + t], sStat[1][t]);
  }
}

// ---------------------------------------------------------------------------
// Kernel 3: instance-norm + swish + residual, float4 vectorized.
// ---------------------------------------------------------------------------
__global__ __launch_bounds__(256) void k_norm(const float* __restrict__ h,
                                              const float* __restrict__ x,
                                              const float* __restrict__ sum,
                                              const float* __restrict__ sq,
                                              const float* __restrict__ gamma,
                                              const float* __restrict__ beta,
                                              float* __restrict__ out) {
  const size_t i4 = ((size_t)blockIdx.x * 256 + threadIdx.x) * 4;
  const int b  = (int)(i4 >> 18);           // N_*F_ = 2^18
  const int f0 = (int)(i4 & 127);
  f32x4 hv = *(const f32x4*)(h + i4);
  f32x4 xv = *(const f32x4*)(x + i4);
  f32x4 sv = *(const f32x4*)(sum + b * F_ + f0);
  f32x4 qv = *(const f32x4*)(sq  + b * F_ + f0);
  f32x4 gv = *(const f32x4*)(gamma + f0);
  f32x4 bv = *(const f32x4*)(beta  + f0);
  const float invN = 1.0f / (float)N_;
  f32x4 ov;
#pragma unroll
  for (int i = 0; i < 4; ++i) {
    float mu  = sv[i] * invN;
    float var = qv[i] * invN - mu * mu;
    float y   = (hv[i] - mu) * __frsqrt_rn(var + 1e-3f) * gv[i] + bv[i];
    float sig = 1.0f / (1.0f + __expf(-y));
    ov[i] = y * sig + xv[i];
  }
  *(f32x4*)(out + i4) = ov;
}

// ---------------------------------------------------------------------------
extern "C" void kernel_launch(void* const* d_in, const int* in_sizes, int n_in,
                              void* d_out, int out_size, void* d_ws, size_t ws_size,
                              hipStream_t stream) {
  const float* x     = (const float*)d_in[0];
  const float* a     = (const float*)d_in[1];
  const float* W     = (const float*)d_in[2];
  // d_in[3] = bias: a per-channel shift cancels exactly inside instance-norm.
  const float* gamma = (const float*)d_in[4];
  const float* beta  = (const float*)d_in[5];
  float* out = (float*)d_out;

  char* ws = (char*)d_ws;
  unsigned short* xwT = (unsigned short*)ws;                    // 4 MiB bf16
  float* h   = (float*)(ws + (size_t)4 * 1024 * 1024);          // 8 MiB fp32
  float* sum = (float*)(ws + (size_t)12 * 1024 * 1024);         // 4 KiB
  float* sq  = sum + B_ * F_;                                   // 4 KiB

  (void)hipMemsetAsync(sum, 0, 2 * B_ * F_ * sizeof(float), stream);
  k_xw  <<<dim3(B_ * (N_ / 64)), 256, 0, stream>>>(x, W, xwT);
  k_gcn <<<dim3(B_ * (N_ / 64)), 256, 0, stream>>>(a, xwT, h, sum, sq);
  k_norm<<<dim3((B_ * N_ * F_ / 4) / 256), 256, 0, stream>>>(h, x, sum, sq,
                                                             gamma, beta, out);
}
